// MoE_1795296330049
// MI455X (gfx1250) — matrix operand order repaired and checked by reading, not compiled, not run on hardware
//
#include <hip/hip_runtime.h>

typedef __attribute__((ext_vector_type(16))) __bf16 bf16x16;
typedef __attribute__((ext_vector_type(2)))  __bf16 bf16x2;
typedef __attribute__((ext_vector_type(8)))  float  v8f;

union FragAB { bf16x16 v; uint4 q[2]; };

#define NTOK   512
#define DDIM   2048
#define FDIM   4096
#define NEXP   8

__device__ __forceinline__ unsigned short f2bf(float f) {
    unsigned int u = __float_as_uint(f);
    unsigned int r = u + 0x7FFFu + ((u >> 16) & 1u);   // RNE
    return (unsigned short)(r >> 16);
}

// pack two f32 -> packed bf16x2 dword (lo in [15:0], hi in [31:16])
#if defined(__has_builtin)
#if __has_builtin(__builtin_amdgcn_cvt_pk_bf16_f32)
#define HAVE_CVT_PK_BF16 1
#endif
#endif
__device__ __forceinline__ unsigned int pack2bf(float lo, float hi) {
#ifdef HAVE_CVT_PK_BF16
    union { bf16x2 v; unsigned int u; } cv;
    cv.v = __builtin_amdgcn_cvt_pk_bf16_f32(lo, hi);
    return cv.u;
#else
    // round-to-nearest (ties away) then merge high halves with one v_perm_b32
    unsigned int ul = __float_as_uint(lo) + 0x8000u;
    unsigned int uh = __float_as_uint(hi) + 0x8000u;
    return __builtin_amdgcn_perm(uh, ul, 0x07060302u);
#endif
}

// ---------------- init: zero out + counters ----------------
__global__ void init_kernel(float* __restrict__ out, int* __restrict__ counts, int n) {
    int i = blockIdx.x * blockDim.x + threadIdx.x;
    if (i < n) out[i] = 0.0f;
    if (i < NEXP) counts[i] = 0;
}

// ---------------- gating: logits, top-2 softmax, routing lists, x->bf16 ----------------
__global__ __launch_bounds__(256) void gate_topk_kernel(
    const float* __restrict__ x, const float* __restrict__ gw,
    int* __restrict__ counts, int* __restrict__ toks, float* __restrict__ wts,
    unsigned short* __restrict__ xbf) {
    const int t    = blockIdx.x;           // token
    const int tid  = threadIdx.x;
    const int e    = tid >> 5;
    const int lane = tid & 31;
    const float* xr = x + (size_t)t * DDIM;

    float p = 0.0f;
    for (int d = lane; d < DDIM; d += 32) p += xr[d] * gw[d * NEXP + e];
    for (int m = 16; m >= 1; m >>= 1) p += __shfl_xor(p, m, 32);

    __shared__ float logit[NEXP];
    if (lane == 0) logit[e] = p;

    // convert this token's row to bf16 (8 elems/thread, packed 16B store)
    const int d0 = tid * 8;
    float4 xa = *(const float4*)(xr + d0);
    float4 xb = *(const float4*)(xr + d0 + 4);
    uint4 pk;
    pk.x = pack2bf(xa.x, xa.y); pk.y = pack2bf(xa.z, xa.w);
    pk.z = pack2bf(xb.x, xb.y); pk.w = pack2bf(xb.z, xb.w);
    *(uint4*)(&xbf[(size_t)t * DDIM + d0]) = pk;

    __syncthreads();
    if (tid == 0) {
        int b1 = 0; float v1 = -3.4e38f;
        for (int i = 0; i < NEXP; ++i) if (logit[i] > v1) { v1 = logit[i]; b1 = i; }
        int b2 = 0; float v2 = -3.4e38f;
        for (int i = 0; i < NEXP; ++i) if (i != b1 && logit[i] > v2) { v2 = logit[i]; b2 = i; }
        float e2 = __expf(v2 - v1);
        float s  = __builtin_amdgcn_rcpf(1.0f + e2);
        int p1 = atomicAdd(&counts[b1], 1);
        toks[b1 * NTOK + p1] = t; wts[b1 * NTOK + p1] = s;
        int p2 = atomicAdd(&counts[b2], 1);
        toks[b2 * NTOK + p2] = t; wts[b2 * NTOK + p2] = e2 * s;
    }
}

// ---------------- prefix offsets ----------------
__global__ void prefix_kernel(const int* __restrict__ counts, int* __restrict__ base) {
    if (threadIdx.x == 0) {
        int s = 0;
        for (int e = 0; e < NEXP; ++e) { base[e] = s; s += counts[e]; }
    }
}

// B-tile staging: pack 8 k-rows x 2 f-cols of f32 into K-contiguous bf16 LDS rows
__device__ __forceinline__ void stage_b(unsigned short* __restrict__ Bt,
                                        const float2 s[8], int fcol, int kb) {
    uint4 d;
    d.x = pack2bf(s[0].x, s[1].x); d.y = pack2bf(s[2].x, s[3].x);
    d.z = pack2bf(s[4].x, s[5].x); d.w = pack2bf(s[6].x, s[7].x);
    *(uint4*)(&Bt[(fcol + 0) * 40 + kb]) = d;
    d.x = pack2bf(s[0].y, s[1].y); d.y = pack2bf(s[2].y, s[3].y);
    d.z = pack2bf(s[4].y, s[5].y); d.w = pack2bf(s[6].y, s[7].y);
    *(uint4*)(&Bt[(fcol + 1) * 40 + kb]) = d;
}

// ---------------- stage A: h = silu(Xg@w1) * (Xg@w3)  [bf16 WMMA, double-buffered] ----------------
// block tile: 64 tokens x 128 F ; 8 waves: 4 in M x 2 in F, each 16x64
__global__ __launch_bounds__(256) void ffn_h_kernel(
    const unsigned short* __restrict__ xbf,
    const float* __restrict__ w1, const float* __restrict__ w3,
    const int* __restrict__ counts, const int* __restrict__ base,
    const int* __restrict__ toks, unsigned short* __restrict__ hbuf) {
    const int e   = blockIdx.z;
    const int cnt = counts[e];
    const int m0  = blockIdx.y * 64;
    if (m0 >= cnt) return;                      // uniform early exit
    const int f0  = blockIdx.x * 128;
    const int tid = threadIdx.x;
    const int lane = tid & 31, wave = tid >> 5;
    const int wm = wave >> 1, wf = wave & 1;

    __shared__ __align__(16) unsigned short At [2 * 64  * 40];
    __shared__ __align__(16) unsigned short B1t[2 * 128 * 40];
    __shared__ __align__(16) unsigned short B3t[2 * 128 * 40];

    v8f zero = {0.f,0.f,0.f,0.f,0.f,0.f,0.f,0.f};
    v8f acc1[4], acc3[4];
    #pragma unroll
    for (int i = 0; i < 4; ++i) { acc1[i] = zero; acc3[i] = zero; }

    const size_t wbase = (size_t)e * DDIM * FDIM + f0;
    const int ar = tid >> 2, ac = tid & 3;              // A loader coords
    const int gt = (m0 + ar < cnt) ? toks[e * NTOK + m0 + ar] : -1;
    const int fcol = (tid & 63) * 2;                    // B loader: f column pair
    const int kb   = (tid >> 6) * 8;                    // B loader: k chunk of 8
    const int am = lane & 15, half = lane >> 4;

    uint4  aS;
    float2 s1[8], s3[8];

    auto loadG = [&](int k0) {
        aS = make_uint4(0u, 0u, 0u, 0u);
        if (gt >= 0) aS = *(const uint4*)(&xbf[(size_t)gt * DDIM + k0 + ac * 8]);
        const float* p1 = w1 + wbase + (size_t)(k0 + kb) * FDIM + fcol;
        const float* p3 = w3 + wbase + (size_t)(k0 + kb) * FDIM + fcol;
        __builtin_prefetch(p1 + (size_t)64 * FDIM, 0, 1);
        __builtin_prefetch(p3 + (size_t)64 * FDIM, 0, 1);
        #pragma unroll
        for (int r = 0; r < 8; ++r) {
            s1[r] = *(const float2*)(p1 + (size_t)r * FDIM);
            s3[r] = *(const float2*)(p3 + (size_t)r * FDIM);
        }
    };
    auto storeL = [&](int buf) {
        *(uint4*)(&At[buf * 2560 + ar * 40 + ac * 8]) = aS;
        stage_b(&B1t[buf * 5120], s1, fcol, kb);
        stage_b(&B3t[buf * 5120], s3, fcol, kb);
    };

    loadG(0);
    storeL(0);
    int cur = 0;
    for (int k0 = 0; k0 < DDIM; k0 += 32) {
        const int nxt = k0 + 32;
        if (nxt < DDIM) loadG(nxt);
        __syncthreads();
        // A fragment: row = wm*16 + (lane%16); K chunks {half*16..+7, half*16+8..+15}
        FragAB fa;
        const unsigned short* Ab = &At[cur * 2560];
        const int arow = wm * 16 + am;
        fa.q[0] = *(const uint4*)(&Ab[arow * 40 + half * 16]);
        fa.q[1] = *(const uint4*)(&Ab[arow * 40 + half * 16 + 8]);
        const unsigned short* B1b = &B1t[cur * 5120];
        const unsigned short* B3b = &B3t[cur * 5120];
        // hoist all B-fragment loads so ds_loads stream ahead of the WMMAs
        FragAB fb1[4], fb3[4];
        #pragma unroll
        for (int nt = 0; nt < 4; ++nt) {
            const int n = wf * 64 + nt * 16 + am;       // local F column
            fb1[nt].q[0] = *(const uint4*)(&B1b[n * 40 + half * 16]);
            fb1[nt].q[1] = *(const uint4*)(&B1b[n * 40 + half * 16 + 8]);
            fb3[nt].q[0] = *(const uint4*)(&B3b[n * 40 + half * 16]);
            fb3[nt].q[1] = *(const uint4*)(&B3b[n * 40 + half * 16 + 8]);
        }
        #pragma unroll
        for (int nt = 0; nt < 4; ++nt) {
            acc1[nt] = __builtin_amdgcn_wmma_f32_16x16x32_bf16(
                false, fa.v, false, fb1[nt].v, (short)0, acc1[nt], false, false);
            acc3[nt] = __builtin_amdgcn_wmma_f32_16x16x32_bf16(
                false, fa.v, false, fb3[nt].v, (short)0, acc3[nt], false, false);
        }
        if (nxt < DDIM) storeL(cur ^ 1);
        cur ^= 1;
    }

    // SwiGLU epilogue -> h (bf16)
    const int be = base[e];
    #pragma unroll
    for (int nt = 0; nt < 4; ++nt) {
        #pragma unroll
        for (int v = 0; v < 8; ++v) {
            const int rl = wm * 16 + half * 8 + v;
            const int grow = m0 + rl;
            if (grow < cnt) {
                float a1 = acc1[nt][v], a3 = acc3[nt][v];
                float hv = a1 * __builtin_amdgcn_rcpf(1.0f + __expf(-a1)) * a3;
                const int col = f0 + wf * 64 + nt * 16 + am;
                hbuf[(size_t)(be + grow) * FDIM + col] = f2bf(hv);
            }
        }
    }
}

// ---------------- stage B: out += wt * (h @ w2)  [bf16 WMMA + f32 atomics] ----------------
__global__ __launch_bounds__(256) void ffn_out_kernel(
    const unsigned short* __restrict__ hbuf, const float* __restrict__ w2,
    const int* __restrict__ counts, const int* __restrict__ base,
    const int* __restrict__ toks, const float* __restrict__ wts,
    float* __restrict__ out) {
    const int e   = blockIdx.z;
    const int cnt = counts[e];
    const int m0  = blockIdx.y * 64;
    if (m0 >= cnt) return;
    const int d0  = blockIdx.x * 128;
    const int tid = threadIdx.x;
    const int lane = tid & 31, wave = tid >> 5;
    const int wm = wave >> 1, wd = wave & 1;

    __shared__ __align__(16) unsigned short At[2 * 64  * 40];
    __shared__ __align__(16) unsigned short Bt[2 * 128 * 40];

    v8f zero = {0.f,0.f,0.f,0.f,0.f,0.f,0.f,0.f};
    v8f acc[4];
    #pragma unroll
    for (int i = 0; i < 4; ++i) acc[i] = zero;

    const int be = base[e];
    const int ar = tid >> 2, ac = tid & 3;
    const bool avalid = (m0 + ar < cnt);
    const int fcol = (tid & 63) * 2;
    const int kb   = (tid >> 6) * 8;
    const int am = lane & 15, half = lane >> 4;

    uint4  aS;
    float2 s2[8];

    auto loadG = [&](int k0) {
        aS = make_uint4(0u, 0u, 0u, 0u);
        if (avalid) aS = *(const uint4*)(&hbuf[(size_t)(be + m0 + ar) * FDIM + k0 + ac * 8]);
        const float* p2 = w2 + ((size_t)e * FDIM + k0 + kb) * DDIM + d0 + fcol;
        __builtin_prefetch(p2 + (size_t)64 * DDIM, 0, 1);
        #pragma unroll
        for (int r = 0; r < 8; ++r) s2[r] = *(const float2*)(p2 + (size_t)r * DDIM);
    };
    auto storeL = [&](int buf) {
        *(uint4*)(&At[buf * 2560 + ar * 40 + ac * 8]) = aS;
        stage_b(&Bt[buf * 5120], s2, fcol, kb);
    };

    loadG(0);
    storeL(0);
    int cur = 0;
    for (int k0 = 0; k0 < FDIM; k0 += 32) {
        const int nxt = k0 + 32;
        if (nxt < FDIM) loadG(nxt);
        __syncthreads();
        FragAB fa;
        const unsigned short* Ab = &At[cur * 2560];
        const int arow = wm * 16 + am;
        fa.q[0] = *(const uint4*)(&Ab[arow * 40 + half * 16]);
        fa.q[1] = *(const uint4*)(&Ab[arow * 40 + half * 16 + 8]);
        const unsigned short* Bb = &Bt[cur * 5120];
        FragAB fb[4];
        #pragma unroll
        for (int nt = 0; nt < 4; ++nt) {
            const int n = wd * 64 + nt * 16 + am;
            fb[nt].q[0] = *(const uint4*)(&Bb[n * 40 + half * 16]);
            fb[nt].q[1] = *(const uint4*)(&Bb[n * 40 + half * 16 + 8]);
        }
        #pragma unroll
        for (int nt = 0; nt < 4; ++nt) {
            acc[nt] = __builtin_amdgcn_wmma_f32_16x16x32_bf16(
                false, fa.v, false, fb[nt].v, (short)0, acc[nt], false, false);
        }
        if (nxt < FDIM) storeL(cur ^ 1);
        cur ^= 1;
    }

    #pragma unroll
    for (int nt = 0; nt < 4; ++nt) {
        #pragma unroll
        for (int v = 0; v < 8; ++v) {
            const int rl = wm * 16 + half * 8 + v;
            const int grow = m0 + rl;
            if (grow < cnt) {
                const float wgt = wts[e * NTOK + grow];
                const int  tok  = toks[e * NTOK + grow];
                const int  col  = d0 + wd * 64 + nt * 16 + am;
                atomicAdd(&out[(size_t)tok * DDIM + col], acc[nt][v] * wgt);
            }
        }
    }
}

// ---------------- launch ----------------
extern "C" void kernel_launch(void* const* d_in, const int* in_sizes, int n_in,
                              void* d_out, int out_size, void* d_ws, size_t ws_size,
                              hipStream_t stream) {
    (void)in_sizes; (void)n_in; (void)out_size; (void)ws_size;
    const float* x  = (const float*)d_in[0];
    const float* gw = (const float*)d_in[1];
    const float* w1 = (const float*)d_in[2];
    const float* w3 = (const float*)d_in[3];
    const float* w2 = (const float*)d_in[4];
    float* out = (float*)d_out;

    char* ws = (char*)d_ws;
    int*            counts = (int*)(ws);                                  // 8 int
    int*            basep  = (int*)(ws + 32);                             // 8 int
    int*            toks   = (int*)(ws + 64);                             // 8*512 int
    float*          wts    = (float*)(ws + 64 + NEXP * NTOK * 4);         // 8*512 f32
    unsigned short* xbf    = (unsigned short*)(ws + 64 + 2 * NEXP * NTOK * 4);
    unsigned short* hbuf   = (unsigned short*)(ws + 64 + 2 * NEXP * NTOK * 4
                                               + (size_t)NTOK * DDIM * 2);

    init_kernel<<<(NTOK * DDIM + 255) / 256, 256, 0, stream>>>(out, counts, NTOK * DDIM);
    gate_topk_kernel<<<NTOK, 256, 0, stream>>>(x, gw, counts, toks, wts, xbf);
    prefix_kernel<<<1, 32, 0, stream>>>(counts, basep);
    ffn_h_kernel<<<dim3(FDIM / 128, NTOK / 64, NEXP), 256, 0, stream>>>(
        xbf, w1, w3, counts, basep, toks, hbuf);
    ffn_out_kernel<<<dim3(DDIM / 128, NTOK / 64, NEXP), 256, 0, stream>>>(
        hbuf, w2, counts, basep, toks, wts, out);
}
